// Classifier_45483703664785
// MI455X (gfx1250) — matrix operand order repaired and checked by reading, not compiled, not run on hardware
//
#include <hip/hip_runtime.h>

typedef __attribute__((ext_vector_type(16))) _Float16 v16h;
typedef __attribute__((ext_vector_type(8)))  float    v8f;

// ---------------------------------------------------------------------------
// Utility kernels
// ---------------------------------------------------------------------------
__global__ void zero_f(float* __restrict__ p, long n) {
    long i = (long)blockIdx.x * blockDim.x + threadIdx.x;
    long stride = (long)gridDim.x * blockDim.x;
    for (; i < n; i += stride) p[i] = 0.0f;
}

__global__ void degree_kernel(const int* __restrict__ src, float* __restrict__ deg, long nE) {
    long i = (long)blockIdx.x * blockDim.x + threadIdx.x;
    if (i < nE) atomicAdd(&deg[src[i]], 1.0f);
}

__global__ void norm_kernel(float* __restrict__ deg, int n) {
    int i = blockIdx.x * blockDim.x + threadIdx.x;
    if (i < n) deg[i] = rsqrtf(fmaxf(deg[i], 1.0f));
}

// One wave per edge: 32 lanes x float4 = 128 channels. Contiguous 512B gather
// from feat[src], contiguous 512B of global_atomic_add_f32 into out[dst].
__global__ void scatter_add128(const int* __restrict__ src, const int* __restrict__ dst,
                               const float* __restrict__ feat, float* __restrict__ out,
                               long nWork) {
    long i = (long)blockIdx.x * blockDim.x + threadIdx.x;
    if (i >= nWork) return;
    long e = i >> 5;
    int  q = (int)(i & 31);
    long s = src[e];
    long d = dst[e];
    const float4 v = ((const float4*)(feat + s * 128))[q];
    float* o = out + d * 128 + (long)q * 4;
    atomicAdd(o + 0, v.x);
    atomicAdd(o + 1, v.y);
    atomicAdd(o + 2, v.z);
    atomicAdd(o + 3, v.w);
}

// h1n = relu(x*norm + b) * norm   (fused layer-1 epilogue + layer-2 pre-scale)
__global__ void epi_prescale_kernel(float* __restrict__ buf, const float* __restrict__ norm,
                                    const float* __restrict__ bias, long total) {
    long i = (long)blockIdx.x * blockDim.x + threadIdx.x;
    if (i >= total) return;
    long row = i >> 7;
    int  c   = (int)(i & 127);
    float nv = norm[row];
    buf[i] = fmaxf(buf[i] * nv + bias[c], 0.0f) * nv;
}

// ---------------------------------------------------------------------------
// WMMA GEMM: out[M x Nout] = op(A[M x K]) @ B, f16 inputs, f32 accumulate.
//   PRESCALE: scale A row i by norm[i] before the GEMM
//   EPI: 0 = none, 1 = relu(acc*norm[row] + bias[col]), 2 = acc + bias[col]
//   TRANSB: B element (k,n) taken from W[n*ldb + k] (for h @ w3^T)
//   DUP: also store at out + M*ldo (tuple output)
// One wave computes one 16x16 tile; 4 waves (blockDim.y) per block over M.
// ---------------------------------------------------------------------------
template <int KTOT, bool PRESCALE, int EPI, bool TRANSB, bool DUP>
__global__ void wmma_gemm(const float* __restrict__ A, const float* __restrict__ B,
                          const float* __restrict__ norm, const float* __restrict__ bias,
                          float* __restrict__ out, int M, int lda, int ldb, int ldo) {
    const int lane  = threadIdx.x;                       // 0..31
    const int mtile = blockIdx.x * blockDim.y + threadIdx.y;
    const int ntile = blockIdx.y;
    if (mtile * 16 >= M) return;                         // wave-uniform exit

    const int rowA = mtile * 16 + (lane & 15);
    const int kA0  = (lane < 16) ? 0 : 8;                // ISA A-frag layout (16-bit, 16x32)
    const int colB = ntile * 16 + (lane & 15);
    const int kB0  = (lane < 16) ? 0 : 8;                // B-frag layout matching A (32x16)

    const float s = PRESCALE ? norm[rowA] : 1.0f;

    v8f acc = {};
    for (int kk = 0; kk < KTOT; kk += 32) {
        v16h a, b;
        const float* ap = A + (long)rowA * lda + kk + kA0;
#pragma unroll
        for (int t = 0; t < 8; ++t) {
            a[t]     = (_Float16)(ap[t] * s);            // K = kk+kA0+t
            a[8 + t] = (_Float16)(ap[16 + t] * s);       // K = kk+kA0+16+t
        }
        if (!TRANSB) {
            const float* bp = B + (long)(kk + kB0) * ldb + colB;
#pragma unroll
            for (int t = 0; t < 16; ++t) b[t] = (_Float16)bp[(long)(t < 8 ? t : t + 8) * ldb];
        } else {
            const float* bp = B + (long)colB * ldb + kk + kB0;
#pragma unroll
            for (int t = 0; t < 16; ++t) b[t] = (_Float16)bp[t < 8 ? t : t + 8];
        }
        acc = __builtin_amdgcn_wmma_f32_16x16x32_f16(
            /*neg_a=*/false, a, /*neg_b=*/false, b,
            /*c_mod=*/(short)0, acc, /*reuse_a=*/false, /*reuse_b=*/false);
    }

    const int col   = ntile * 16 + (lane & 15);
    const int rbase = mtile * 16 + ((lane < 16) ? 0 : 8);
#pragma unroll
    for (int r = 0; r < 8; ++r) {
        const int row = rbase + r;
        float v = acc[r];
        if (EPI == 1)      v = fmaxf(v * norm[row] + bias[col], 0.0f);
        else if (EPI == 2) v = v + bias[col];
        out[(long)row * ldo + col] = v;
        if (DUP) out[(long)(row + M) * ldo + col] = v;
    }
}

// ---------------------------------------------------------------------------
extern "C" void kernel_launch(void* const* d_in, const int* in_sizes, int n_in,
                              void* d_out, int out_size, void* d_ws, size_t ws_size,
                              hipStream_t stream) {
    const int    M      = in_sizes[0];            // 100000 nodes
    const long   nEdges = (long)in_sizes[1] / 2;  // 1.6M edges
    const int*   e_src  = (const int*)d_in[1];
    const int*   e_dst  = e_src + nEdges;
    const float* feats  = (const float*)d_in[3];  // [M,256]
    const float* w1     = (const float*)d_in[4];  // [256,128]
    const float* b1     = (const float*)d_in[5];
    const float* w2     = (const float*)d_in[6];  // [128,128]
    const float* b2     = (const float*)d_in[7];
    const float* w3     = (const float*)d_in[8];  // [16,128]
    const float* b3     = (const float*)d_in[9];
    float*       outp   = (float*)d_out;

    // Workspace: norm[M] | bufA[M*128] | bufB[M*128]  (~103 MB)
    float* normb = (float*)d_ws;
    float* bufA  = normb + (((long)M + 255) / 256) * 256;
    float* bufB  = bufA + (long)M * 128;

    const long mc128   = (long)M * 128;
    const long scatterW = nEdges * 32;

    // 1) degree -> norm = rsqrt(max(deg,1))
    zero_f<<<1024, 256, 0, stream>>>(normb, M);
    degree_kernel<<<(int)((nEdges + 255) / 256), 256, 0, stream>>>(e_src, normb, nEdges);
    norm_kernel<<<(M + 255) / 256, 256, 0, stream>>>(normb, M);

    const int mtiles = (M + 15) / 16;               // 6250
    dim3 gblock(32, 4);
    dim3 ggrid((mtiles + 3) / 4, 128 / 16);

    // 2) Layer 1: bufA = (feats*norm) @ w1   (WMMA, K=256)
    wmma_gemm<256, true, 0, false, false><<<ggrid, gblock, 0, stream>>>(
        feats, w1, normb, nullptr, bufA, M, 256, 128, 128);

    // 3) bufB = segment_sum over edges of bufA[src]
    zero_f<<<4096, 256, 0, stream>>>(bufB, mc128);
    scatter_add128<<<(int)((scatterW + 255) / 256), 256, 0, stream>>>(e_src, e_dst, bufA, bufB, scatterW);

    // 4) bufB = relu(bufB*norm + b1) * norm   (layer-1 epilogue fused with layer-2 pre-scale)
    epi_prescale_kernel<<<(int)((mc128 + 255) / 256), 256, 0, stream>>>(bufB, normb, b1, mc128);

    // 5) bufA = segment_sum over edges of bufB[src]
    zero_f<<<4096, 256, 0, stream>>>(bufA, mc128);
    scatter_add128<<<(int)((scatterW + 255) / 256), 256, 0, stream>>>(e_src, e_dst, bufB, bufA, scatterW);

    // 6) Layer 2: bufB = relu((bufA @ w2)*norm + b2)   (WMMA, K=128, fused epilogue)
    wmma_gemm<128, false, 1, false, false><<<ggrid, gblock, 0, stream>>>(
        bufA, w2, normb, b2, bufB, M, 128, 128, 128);

    // 7) out = bufB @ w3^T + b3, stored twice (tuple)   (WMMA, N=16, B transposed)
    dim3 ggrid3((mtiles + 3) / 4, 1);
    wmma_gemm<128, false, 2, true, true><<<ggrid3, gblock, 0, stream>>>(
        bufB, w3, normb, b3, outp, M, 128, 128, 16);
}
